// MultiLayerCrossModalAttention_16466904613744
// MI455X (gfx1250) — matrix operand order repaired and checked by reading, not compiled, not run on hardware
//
#include <hip/hip_runtime.h>
#include <hip/hip_bf16.h>

#define Cch   128
#define HW    16384            // 128*128 pixels per (batch, channel) plane
#define NB    4                // batches
#define NTOK  32               // token grid is 32x32 (TS=4)
#define SCALE 0.25f            // (TS*TS)^-0.5
#define LNEPS 1e-5f

typedef __attribute__((ext_vector_type(16))) __bf16 v16bf;
typedef __attribute__((ext_vector_type(8)))  __bf16 v8bf;
typedef __attribute__((ext_vector_type(8)))  float  v8f;
typedef __attribute__((ext_vector_type(4)))  unsigned u32x4;
typedef __attribute__((ext_vector_type(8)))  unsigned u32x8;

__device__ __forceinline__ __bf16 f2bf(float f) {
    unsigned u = __builtin_bit_cast(unsigned, f);
    u += 0x7FFFu + ((u >> 16) & 1u);          // round-to-nearest-even
    unsigned short h = (unsigned short)(u >> 16);
    return __builtin_bit_cast(__bf16, h);
}

// ---------------------------------------------------------------------------
// init: d_out = blue, cur_blue = blue
// ---------------------------------------------------------------------------
__global__ void init_kernel(const float* __restrict__ blue,
                            float* __restrict__ out,
                            float* __restrict__ cur, int n) {
    int i = blockIdx.x * blockDim.x + threadIdx.x;
    if (i < n) { float v = blue[i]; out[i] = v; cur[i] = v; }
}

// ---------------------------------------------------------------------------
// cvt_bf16: elementwise fp32 -> bf16 (for the conv weights)
// ---------------------------------------------------------------------------
__global__ void cvt_bf16_kernel(const float* __restrict__ in,
                                __bf16* __restrict__ out, int n) {
    int i = blockIdx.x * blockDim.x + threadIdx.x;
    if (i < n) out[i] = f2bf(in[i]);
}

// ---------------------------------------------------------------------------
// cvt_transpose: X [NB][C][HW] fp32 -> XT [NB][HW][C] bf16  (32x32 LDS tiles)
// grid (HW/32, C/32, NB), block (32,8)
// ---------------------------------------------------------------------------
__global__ void __launch_bounds__(256)
cvt_transpose_kernel(const float* __restrict__ X, __bf16* __restrict__ XT) {
    __shared__ float tile[32][33];
    const int b  = blockIdx.z;
    const int n0 = blockIdx.x * 32;
    const int k0 = blockIdx.y * 32;
    const float* Xb = X  + (size_t)b * Cch * HW;
    __bf16*      Tb = XT + (size_t)b * HW * Cch;
    #pragma unroll
    for (int r = 0; r < 4; ++r) {
        int k = threadIdx.y * 4 + r;
        tile[k][threadIdx.x] = Xb[(size_t)(k0 + k) * HW + n0 + threadIdx.x];
    }
    __syncthreads();
    #pragma unroll
    for (int r = 0; r < 4; ++r) {
        int n = threadIdx.y * 4 + r;
        Tb[(size_t)(n0 + n) * Cch + k0 + threadIdx.x] = f2bf(tile[threadIdx.x][n]);
    }
}

// ---------------------------------------------------------------------------
// gemm_qkv: Out[b][o][n] = sum_c W[o][c] * X[b][c][n] + bias[o]
// X supplied pre-transposed + bf16:  XT[b][n][k].
// grid = (HW/64, NB, 3), block = 256 (8 waves). blockIdx.z selects Q/K/V.
// TDM stages the [64 n][128 k] bf16 tile into LDS (wave 0 issues, all sync).
// Each wave: one 16-row M tile, 4 N subtiles, K=128 in 4 WMMA steps.
// ---------------------------------------------------------------------------
__global__ void __launch_bounds__(256)
gemm_qkv_kernel(const __bf16* __restrict__ curT, const __bf16* __restrict__ whiteT,
                const __bf16* __restrict__ wq, const __bf16* __restrict__ wk,
                const __bf16* __restrict__ wv,
                const float* __restrict__ qb, const float* __restrict__ kb,
                const float* __restrict__ vb,
                float* __restrict__ Q, float* __restrict__ K, float* __restrict__ V) {
    const int which = blockIdx.z;
    const __bf16* XT = (which == 0) ? curT : whiteT;
    const __bf16* Wm = (which == 0) ? wq : (which == 1 ? wk : wv);
    const float*  Bv = (which == 0) ? qb : (which == 1 ? kb : vb);
    float*        Ot = (which == 0) ? Q  : (which == 1 ? K  : V);

    const int bb = blockIdx.y;
    const int n0 = blockIdx.x * 64;
    float* Ob = Ot + (size_t)bb * (Cch * HW);

    __shared__ __bf16 ldsX[64 * 128];        // packed [n][k] bf16, 16 KB

    if (threadIdx.x < 32) {
        // ---- Tensor Data Mover: 2D tile [dim1=64 rows (n)] x [dim0=128 (k)] ----
        unsigned long long gaddr = (unsigned long long)(uintptr_t)XT
                                 + ((size_t)bb * HW + n0) * Cch * 2ull;
        unsigned ldsa = (unsigned)(uintptr_t)(void*)ldsX;   // addr[31:0] == LDS offset
        u32x4 g0;
        g0.x = 1u;                                          // count=1, user desc
        g0.y = ldsa;                                        // lds_addr
        g0.z = (unsigned)(gaddr & 0xFFFFFFFFull);           // global_addr[31:0]
        g0.w = (unsigned)((gaddr >> 32) & 0x01FFFFFFull)    // global_addr[56:32]
             | 0x80000000u;                                 // type=2 (image)
        u32x8 g1;
        g1.s0 = 0x00010000u;          // wg_mask=0, data_size=1 (2B), no pad/iter
        g1.s1 = (unsigned)Cch << 16;  // tensor_dim0[15:0]=128 in bits 63:48
        g1.s2 = (unsigned)HW  << 16;  // tensor_dim1[15:0]=16384 in bits 95:80
        g1.s3 = (unsigned)Cch << 16;  // tile_dim0=128 in bits 127:112
        g1.s4 = 64u;                  // tile_dim1=64 (tile_dim2=0)
        g1.s5 = (unsigned)Cch;        // tensor_dim0_stride=128
        g1.s6 = 0u;
        g1.s7 = 0u;
        asm volatile("tensor_load_to_lds %0, %1, null, null"
                     :: "s"(g0), "s"(g1) : "memory");
        __builtin_amdgcn_s_wait_tensorcnt(0);
    }
    __syncthreads();

    const int wave = threadIdx.x >> 5, lane = threadIdx.x & 31;
    const int hh = lane >> 4, lm = lane & 15;
    const int row = wave * 16 + lm;            // output-channel row of A

    // A fragments: one contiguous 16-K run per lane (K = kb + 16*hh + s), matching B.
    const __bf16* wrow = Wm + row * Cch;
    v16bf a[4];
    #pragma unroll
    for (int kk = 0; kk < 4; ++kk) {
        a[kk] = *(const v16bf*)(wrow + kk * 32 + hh * 16);
    }

    #pragma unroll
    for (int nt = 0; nt < 4; ++nt) {
        const int nb = nt * 16;
        v8f c = {0.f, 0.f, 0.f, 0.f, 0.f, 0.f, 0.f, 0.f};
        #pragma unroll
        for (int kk = 0; kk < 4; ++kk) {
            // B layout: lanes 0-15 K=kb..kb+15, lanes 16-31 K=kb+16..kb+31, N=lm
            // -> 32 contiguous bytes in the [n][k]-packed LDS tile.
            const __bf16* bp = ldsX + (nb + lm) * Cch + kk * 32 + hh * 16;
            v16bf bm = *(const v16bf*)bp;
            c = __builtin_amdgcn_wmma_f32_16x16x32_bf16(
                    false, a[kk], false, bm, (short)0, c, false, false);
        }
        // C/D f32 layout: VGPR r -> M = r + 8*hh, N = lm
        #pragma unroll
        for (int r = 0; r < 8; ++r) {
            int o = wave * 16 + hh * 8 + r;
            Ob[(size_t)o * HW + n0 + nb + lm] = c[r] + Bv[o];
        }
    }
}

// ---------------------------------------------------------------------------
// attn: per (b, c, token) local 3x3-neighbor attention over D=16.
// grid = (C/8, NTOK, NB), block = (32 tw, 8 c). enh overwrites Q in place.
// ---------------------------------------------------------------------------
__global__ void __launch_bounds__(256)
attn_kernel(float* __restrict__ Qe, const float* __restrict__ Kt,
            const float* __restrict__ Vt) {
    const int tw = threadIdx.x;
    const int c  = blockIdx.x * 8 + threadIdx.y;
    const int th = blockIdx.y;
    const int b  = blockIdx.z;
    const size_t plane = ((size_t)b * Cch + c) * HW;
    const size_t tok0  = plane + (size_t)(th * 4) * 128 + tw * 4;

    float q[16];
    #pragma unroll
    for (int i = 0; i < 4; ++i) {
        float4 r = *(const float4*)(Qe + tok0 + i * 128);
        q[4*i+0] = r.x; q[4*i+1] = r.y; q[4*i+2] = r.z; q[4*i+3] = r.w;
    }

    float logit[9];
    #pragma unroll
    for (int nb = 0; nb < 9; ++nb) {
        const int t2h = th + nb / 3 - 1, t2w = tw + nb % 3 - 1;
        float acc = 0.f;
        if ((unsigned)t2h < (unsigned)NTOK && (unsigned)t2w < (unsigned)NTOK) {
            const size_t base = plane + (size_t)(t2h * 4) * 128 + t2w * 4;
            #pragma unroll
            for (int i = 0; i < 4; ++i) {
                float4 r = *(const float4*)(Kt + base + i * 128);
                acc += q[4*i+0]*r.x + q[4*i+1]*r.y + q[4*i+2]*r.z + q[4*i+3]*r.w;
            }
        }
        logit[nb] = acc * SCALE;   // OOB -> 0, matching zero-pad reference
    }

    float m = logit[0];
    #pragma unroll
    for (int nb = 1; nb < 9; ++nb) m = fmaxf(m, logit[nb]);
    float w[9], s = 0.f;
    #pragma unroll
    for (int nb = 0; nb < 9; ++nb) { w[nb] = __expf(logit[nb] - m); s += w[nb]; }
    const float inv = 1.f / s;

    float o[16];
    #pragma unroll
    for (int d = 0; d < 16; ++d) o[d] = 0.f;
    #pragma unroll
    for (int nb = 0; nb < 9; ++nb) {
        const int t2h = th + nb / 3 - 1, t2w = tw + nb % 3 - 1;
        if ((unsigned)t2h < (unsigned)NTOK && (unsigned)t2w < (unsigned)NTOK) {
            const float ww = w[nb] * inv;
            const size_t base = plane + (size_t)(t2h * 4) * 128 + t2w * 4;
            #pragma unroll
            for (int i = 0; i < 4; ++i) {
                float4 r = *(const float4*)(Vt + base + i * 128);
                o[4*i+0] += ww * r.x; o[4*i+1] += ww * r.y;
                o[4*i+2] += ww * r.z; o[4*i+3] += ww * r.w;
            }
        }
    }
    #pragma unroll
    for (int i = 0; i < 4; ++i) {
        float4 r; r.x = o[4*i+0]; r.y = o[4*i+1]; r.z = o[4*i+2]; r.w = o[4*i+3];
        *(float4*)(Qe + tok0 + i * 128) = r;
    }
}

// ---------------------------------------------------------------------------
// ln_res: channel LayerNorm per pixel + residual + weighted accumulation.
// ---------------------------------------------------------------------------
__global__ void __launch_bounds__(256)
ln_res_kernel(const float* __restrict__ Enh, const float* __restrict__ g,
              const float* __restrict__ bta, const float* __restrict__ lwts,
              int layer, float* __restrict__ cur, float* __restrict__ out) {
    const int p  = blockIdx.x * blockDim.x + threadIdx.x; // 0..65535
    const int b  = p >> 14;
    const int hw = p & (HW - 1);
    const size_t base = (size_t)b * Cch * HW + hw;

    float s = 0.f, s2 = 0.f;
    for (int c2 = 0; c2 < Cch; ++c2) {
        float x = Enh[base + (size_t)c2 * HW];
        s += x; s2 += x * x;
    }
    const float mu  = s  * (1.f / Cch);
    const float var = s2 * (1.f / Cch) - mu * mu;
    const float rs  = rsqrtf(var + LNEPS);
    const float lw  = lwts[layer];

    for (int c2 = 0; c2 < Cch; ++c2) {
        const size_t idx = base + (size_t)c2 * HW;
        float ln = (Enh[idx] - mu) * rs * g[c2] + bta[c2];
        cur[idx] += ln;
        out[idx] += lw * ln;
    }
}

// ---------------------------------------------------------------------------
extern "C" void kernel_launch(void* const* d_in, const int* in_sizes, int n_in,
                              void* d_out, int out_size, void* d_ws, size_t ws_size,
                              hipStream_t stream) {
    const float* blue  = (const float*)d_in[0];
    const float* white = (const float*)d_in[1];
    const float* qw = (const float*)d_in[2];
    const float* qb = (const float*)d_in[3];
    const float* kw = (const float*)d_in[4];
    const float* kb = (const float*)d_in[5];
    const float* vw = (const float*)d_in[6];
    const float* vb = (const float*)d_in[7];
    const float* lng = (const float*)d_in[8];
    const float* lnb = (const float*)d_in[9];
    const float* lwt = (const float*)d_in[10];
    float* out = (float*)d_out;

    const size_t TOT = (size_t)NB * Cch * HW;       // 8,388,608 elements
    float* ws   = (float*)d_ws;
    float* Qb   = ws;                                // fp32 planes
    float* Kb   = ws + TOT;
    float* Vb   = ws + 2 * TOT;
    float* cur  = ws + 3 * TOT;
    __bf16* whiteT = (__bf16*)(ws + 4 * TOT);        // bf16 transposed activations
    __bf16* curT   = whiteT + TOT;
    __bf16* wqBF   = curT + TOT;                     // bf16 weights, 2 layers each
    __bf16* wkBF   = wqBF + 2 * Cch * Cch;
    __bf16* wvBF   = wkBF + 2 * Cch * Cch;

    init_kernel<<<(int)(TOT / 256), 256, 0, stream>>>(blue, out, cur, (int)TOT);

    const int nw = 2 * Cch * Cch;                    // 32768 per weight tensor
    cvt_bf16_kernel<<<nw / 256, 256, 0, stream>>>(qw, wqBF, nw);
    cvt_bf16_kernel<<<nw / 256, 256, 0, stream>>>(kw, wkBF, nw);
    cvt_bf16_kernel<<<nw / 256, 256, 0, stream>>>(vw, wvBF, nw);

    cvt_transpose_kernel<<<dim3(HW / 32, Cch / 32, NB), dim3(32, 8, 1), 0, stream>>>(
        white, whiteT);

    for (int l = 0; l < 2; ++l) {
        cvt_transpose_kernel<<<dim3(HW / 32, Cch / 32, NB), dim3(32, 8, 1), 0, stream>>>(
            cur, curT);
        gemm_qkv_kernel<<<dim3(HW / 64, NB, 3), 256, 0, stream>>>(
            curT, whiteT,
            wqBF + (size_t)l * Cch * Cch, wkBF + (size_t)l * Cch * Cch,
            wvBF + (size_t)l * Cch * Cch,
            qb + l * Cch, kb + l * Cch, vb + l * Cch,
            Qb, Kb, Vb);
        attn_kernel<<<dim3(Cch / 8, NTOK, NB), dim3(32, 8, 1), 0, stream>>>(Qb, Kb, Vb);
        ln_res_kernel<<<256, 256, 0, stream>>>(Qb, lng + l * Cch, lnb + l * Cch,
                                               lwt, l, cur, out);
    }
}
